// MobileNetV2_65257733096092
// MI455X (gfx1250) — 8-bit matrix operand order repaired and checked by reading, compiled, not run on hardware
//
#include <hip/hip_runtime.h>
#include <stdint.h>

typedef __attribute__((ext_vector_type(8))) int v8i;
typedef _Float16 h16;

static __device__ __forceinline__ float dqs(const float* __restrict__ am) {
    return am[0] * (1.0f / 7.0f) + 1e-8f;
}

static __device__ __forceinline__ v8i load_a(const int8_t* __restrict__ p) {
    int2 a0 = *(const int2*)(p + 0);
    int2 a1 = *(const int2*)(p + 16);
    int2 a2 = *(const int2*)(p + 32);
    int2 a3 = *(const int2*)(p + 48);
    v8i v = {a0.x, a0.y, a1.x, a1.y, a2.x, a2.y, a3.x, a3.y};
    return v;
}
static __device__ __forceinline__ v8i load_b(const int8_t* __restrict__ p) {
    int2 b0 = *(const int2*)(p + 0);
    int2 b1 = *(const int2*)(p + 16);
    int2 b2 = *(const int2*)(p + 32);
    int2 b3 = *(const int2*)(p + 48);
    v8i v = {b0.x, b0.y, b1.x, b1.y, b2.x, b2.y, b3.x, b3.y};
    return v;
}

static __device__ __forceinline__ void store_tile(const v8i& acc, h16* __restrict__ O,
                                                  const h16* __restrict__ R, int ldO,
                                                  int ldR, int m0, int n, float s,
                                                  int relu) {
#pragma unroll
    for (int r = 0; r < 8; ++r) {
        float f = (float)acc[r] * s;
        if (R) f += (float)R[(size_t)(m0 + r) * (size_t)ldR + (size_t)n];
        if (relu) f = f > 0.0f ? f : 0.0f;
        O[(size_t)(m0 + r) * (size_t)ldO + (size_t)n] = (h16)f;
    }
}

__global__ void k_amax_f32(const float* __restrict__ p, long long n,
                           unsigned* __restrict__ out) {
    __shared__ float red[256];
    float m = 0.0f;
    for (long long i = (long long)blockIdx.x * blockDim.x + threadIdx.x; i < n;
         i += (long long)gridDim.x * blockDim.x) {
        float v = fabsf(p[i]);
        m = v > m ? v : m;
    }
    red[threadIdx.x] = m;
    __syncthreads();
    for (int s = 128; s > 0; s >>= 1) {
        if ((int)threadIdx.x < s) {
            float o = red[threadIdx.x + s];
            if (o > red[threadIdx.x]) red[threadIdx.x] = o;
        }
        __syncthreads();
    }
    if (threadIdx.x == 0) atomicMax(out, __float_as_uint(red[0]));
}

__global__ void k_amax_f16(const h16* __restrict__ p, long long n,
                           unsigned* __restrict__ out) {
    __shared__ float red[256];
    float m = 0.0f;
    for (long long i = (long long)blockIdx.x * blockDim.x + threadIdx.x; i < n;
         i += (long long)gridDim.x * blockDim.x) {
        float v = fabsf((float)p[i]);
        m = v > m ? v : m;
    }
    red[threadIdx.x] = m;
    __syncthreads();
    for (int s = 128; s > 0; s >>= 1) {
        if ((int)threadIdx.x < s) {
            float o = red[threadIdx.x + s];
            if (o > red[threadIdx.x]) red[threadIdx.x] = o;
        }
        __syncthreads();
    }
    if (threadIdx.x == 0) atomicMax(out, __float_as_uint(red[0]));
}

__global__ void k_packw(const float* __restrict__ W, int N, int K,
                        int8_t* __restrict__ Q, int Np, int Kp,
                        const float* __restrict__ am) {
    long long idx = (long long)blockIdx.x * blockDim.x + threadIdx.x;
    long long tot = (long long)Np * Kp;
    if (idx >= tot) return;
    int k = (int)(idx % Kp);
    int n = (int)(idx / Kp);
    int8_t q = 0;
    if (n < N && k < K) {
        float s = dqs(am);
        float r = rintf(W[(long long)n * K + k] / s);
        r = r > 7.0f ? 7.0f : (r < -7.0f ? -7.0f : r);
        q = (int8_t)(int)r;
    }
    Q[idx] = q;
}

__global__ void k_quant(const h16* __restrict__ F, long long M, int C, int ldF,
                        int8_t* __restrict__ Q, int ldQ,
                        const float* __restrict__ am) {
    long long idx = (long long)blockIdx.x * blockDim.x + threadIdx.x;
    long long tot = M * (long long)ldQ;
    if (idx >= tot) return;
    int c = (int)(idx % ldQ);
    long long m = idx / ldQ;
    int8_t q = 0;
    if (c < C) {
        float s = dqs(am);
        float r = rintf((float)F[m * ldF + c] / s);
        r = r > 7.0f ? 7.0f : (r < -7.0f ? -7.0f : r);
        q = (int8_t)(int)r;
    }
    Q[idx] = q;
}

__global__ void k_gemm_iu8(const int8_t* __restrict__ A, const int8_t* __restrict__ B,
                           h16* __restrict__ O, const h16* __restrict__ R,
                           int M, int N32, int K64, int ldO, int ldR,
                           const float* __restrict__ amA, const float* __restrict__ amB,
                           int relu) {
    int wave = (int)((blockIdx.x * blockDim.x + threadIdx.x) >> 5);
    int tilesM = M >> 5;
    int tilesN = N32 >> 5;
    if (wave >= tilesM * tilesN) return;
    int tm = wave % tilesM;
    int tn = wave / tilesM;
    int lane = (int)(threadIdx.x & 31u);
    int hi = lane >> 4;
    int l16 = lane & 15;

    const int8_t* a0p = A + (size_t)(tm * 32 + l16) * (size_t)K64 + (size_t)(hi * 8);
    const int8_t* a1p = a0p + (size_t)16 * (size_t)K64;
    const int8_t* b0p = B + (size_t)(tn * 32 + l16) * (size_t)K64 + (size_t)(hi * 8);
    const int8_t* b1p = b0p + (size_t)16 * (size_t)K64;

    v8i acc00 = {0, 0, 0, 0, 0, 0, 0, 0};
    v8i acc01 = {0, 0, 0, 0, 0, 0, 0, 0};
    v8i acc10 = {0, 0, 0, 0, 0, 0, 0, 0};
    v8i acc11 = {0, 0, 0, 0, 0, 0, 0, 0};
    int kt = K64 >> 6;
#pragma unroll 2
    for (int k = 0; k < kt; ++k) {
        v8i va0 = load_a(a0p);
        v8i va1 = load_a(a1p);
        v8i vb0 = load_b(b0p);
        v8i vb1 = load_b(b1p);
        __builtin_prefetch(a0p + 64, 0, 3);
        __builtin_prefetch(a1p + 64, 0, 3);
        __builtin_prefetch(b0p + 64, 0, 3);
        __builtin_prefetch(b1p + 64, 0, 3);
        acc00 = __builtin_amdgcn_wmma_i32_16x16x64_iu8(true, va0, true, vb0, acc00,
                                                       false, false);
        acc01 = __builtin_amdgcn_wmma_i32_16x16x64_iu8(true, va0, true, vb1, acc01,
                                                       false, false);
        acc10 = __builtin_amdgcn_wmma_i32_16x16x64_iu8(true, va1, true, vb0, acc10,
                                                       false, false);
        acc11 = __builtin_amdgcn_wmma_i32_16x16x64_iu8(true, va1, true, vb1, acc11,
                                                       false, false);
        a0p += 64; a1p += 64; b0p += 64; b1p += 64;
    }
    float s = dqs(amA) * dqs(amB);
    int n0 = tn * 32 + l16;
    int m0 = tm * 32 + hi * 8;
    store_tile(acc00, O, R, ldO, ldR, m0 + 0,  n0 + 0,  s, relu);
    store_tile(acc01, O, R, ldO, ldR, m0 + 0,  n0 + 16, s, relu);
    store_tile(acc10, O, R, ldO, ldR, m0 + 16, n0 + 0,  s, relu);
    store_tile(acc11, O, R, ldO, ldR, m0 + 16, n0 + 16, s, relu);
}

__global__ void k_dwconv(const int8_t* __restrict__ X, const int8_t* __restrict__ W9,
                         h16* __restrict__ O, int Hin, int C, int Cp, int ldX,
                         int stride, const float* __restrict__ amX,
                         const float* __restrict__ amW) {
    int Hout = Hin / stride;
    long long idx = (long long)blockIdx.x * blockDim.x + threadIdx.x;
    long long tot = 128LL * Hout * Hout * Cp;
    if (idx >= tot) return;
    int c = (int)(idx % Cp);
    long long p = idx / Cp;
    int wo = (int)(p % Hout);
    p /= Hout;
    int ho = (int)(p % Hout);
    long long b = p / Hout;
    float v = 0.0f;
    if (c < C) {
        int acc = 0;
        int hi0 = ho * stride - 1, wi0 = wo * stride - 1;
#pragma unroll
        for (int kh = 0; kh < 3; ++kh) {
            int hh = hi0 + kh;
            if ((unsigned)hh < (unsigned)Hin) {
#pragma unroll
                for (int kw = 0; kw < 3; ++kw) {
                    int ww = wi0 + kw;
                    if ((unsigned)ww < (unsigned)Hin) {
                        int xv = X[((b * Hin + hh) * Hin + ww) * (long long)ldX + c];
                        int wv = W9[c * 9 + kh * 3 + kw];
                        acc += xv * wv;
                    }
                }
            }
        }
        v = (float)acc * dqs(amX) * dqs(amW);
        v = v > 0.0f ? v : 0.0f;
    }
    O[idx] = (h16)v;
}

__global__ void k_conv1(const float* __restrict__ X, const int8_t* __restrict__ Wq,
                        const float* __restrict__ amW, h16* __restrict__ O) {
    int idx = (int)(blockIdx.x * blockDim.x + threadIdx.x);
    if (idx >= 128 * 32 * 32 * 32) return;
    int co = idx & 31;
    int p = idx >> 5;
    int w = p & 31;
    int h = (p >> 5) & 31;
    int b = p >> 10;
    float acc = 0.0f;
#pragma unroll
    for (int ci = 0; ci < 3; ++ci) {
#pragma unroll
        for (int kh = 0; kh < 3; ++kh) {
            int hh = h + kh - 1;
            if ((unsigned)hh < 32u) {
#pragma unroll
                for (int kw = 0; kw < 3; ++kw) {
                    int ww = w + kw - 1;
                    if ((unsigned)ww < 32u) {
                        acc += X[((b * 3 + ci) * 32 + hh) * 32 + ww] *
                               (float)Wq[co * 27 + ci * 9 + kh * 3 + kw];
                    }
                }
            }
        }
    }
    acc *= dqs(amW);
    O[idx] = (h16)(acc > 0.0f ? acc : 0.0f);
}

__global__ void k_poolfc(const h16* __restrict__ F, const float* __restrict__ fw,
                         const float* __restrict__ fb, float* __restrict__ out) {
    __shared__ float red[10][256];
    int b = blockIdx.x, t = threadIdx.x;
    float part[10];
#pragma unroll
    for (int j = 0; j < 10; ++j) part[j] = 0.0f;
    for (int c = t; c < 1280; c += 256) {
        float mean = 0.0f;
#pragma unroll
        for (int p = 0; p < 16; ++p)
            mean += (float)F[(size_t)(b * 16 + p) * 1280 + c];
        mean *= (1.0f / 16.0f);
#pragma unroll
        for (int j = 0; j < 10; ++j) part[j] += mean * fw[j * 1280 + c];
    }
#pragma unroll
    for (int j = 0; j < 10; ++j) red[j][t] = part[j];
    __syncthreads();
    for (int s = 128; s > 0; s >>= 1) {
        if (t < s)
#pragma unroll
            for (int j = 0; j < 10; ++j) red[j][t] += red[j][t + s];
        __syncthreads();
    }
    if (t < 10) out[b * 10 + t] = red[t][0] + fb[t];
}

static const int SPEC[17][4] = {
    {32, 16, 1, 1},  {16, 24, 6, 1},  {24, 24, 6, 1},  {24, 32, 6, 2},
    {32, 32, 6, 1},  {32, 32, 6, 1},  {32, 64, 6, 2},  {64, 64, 6, 1},
    {64, 64, 6, 1},  {64, 64, 6, 1},  {64, 96, 6, 1},  {96, 96, 6, 1},
    {96, 96, 6, 1},  {96, 160, 6, 2}, {160, 160, 6, 1}, {160, 160, 6, 1},
    {160, 320, 6, 1}};

static inline int p32(int x) { return (x + 31) & ~31; }
static inline int p64(int x) { return (x + 63) & ~63; }

extern "C" void kernel_launch(void* const* d_in, const int* in_sizes, int n_in,
                              void* d_out, int out_size, void* d_ws, size_t ws_size,
                              hipStream_t stream) {
    (void)in_sizes; (void)n_in; (void)out_size; (void)ws_size;
    int ii = 0;
    const float* X = (const float*)d_in[ii++];
    const float* Wc1 = (const float*)d_in[ii++];
    const float *w1f[17], *w2f[17], *w3f[17], *wsf[17];
    for (int i = 0; i < 17; ++i) {
        int inp = SPEC[i][0], outp = SPEC[i][1], st = SPEC[i][3];
        w1f[i] = (const float*)d_in[ii++];
        w2f[i] = (const float*)d_in[ii++];
        w3f[i] = (const float*)d_in[ii++];
        wsf[i] = (st == 1 && inp != outp) ? (const float*)d_in[ii++] : nullptr;
    }
    const float* Wc2 = (const float*)d_in[ii++];
    const float* FCW = (const float*)d_in[ii++];
    const float* FCB = (const float*)d_in[ii++];

    char* base = (char*)d_ws;
    size_t cur = 0;
    auto alloc = [&](size_t b) -> void* {
        cur = (cur + 255) & ~(size_t)255;
        void* p = base + cur;
        cur += b;
        return p;
    };

    float* scal = (float*)alloc(1024);
    int sidx = 0;
    auto slot = [&]() { return scal + (sidx++); };
    hipMemsetAsync(scal, 0, 1024, stream);

    auto amax32 = [&](const float* p, long long n, float* am) {
        long long bl = (n + 255) / 256; if (bl > 4096) bl = 4096;
        k_amax_f32<<<dim3((unsigned)bl), dim3(256), 0, stream>>>(p, n, (unsigned*)am);
    };
    auto amax16 = [&](const h16* p, long long n, float* am) {
        long long bl = (n + 255) / 256; if (bl > 4096) bl = 4096;
        k_amax_f16<<<dim3((unsigned)bl), dim3(256), 0, stream>>>(p, n, (unsigned*)am);
    };
    auto packW = [&](const float* W, int N, int K, int Np, int Kp, int8_t* Q,
                     float* am) {
        amax32(W, (long long)N * K, am);
        long long tot = (long long)Np * Kp;
        k_packw<<<dim3((unsigned)((tot + 255) / 256)), dim3(256), 0, stream>>>(
            W, N, K, Q, Np, Kp, am);
    };

    int8_t* qc1 = (int8_t*)alloc(32 * 27);
    float* ac1 = slot();
    packW(Wc1, 32, 27, 32, 27, qc1, ac1);

    int8_t *qw1[17], *qw2[17], *qw3[17], *qws[17];
    float *aw1[17], *aw2[17], *aw3[17], *aws[17];
    for (int i = 0; i < 17; ++i) {
        int inp = SPEC[i][0], outp = SPEC[i][1], ex = SPEC[i][2];
        int pl = ex * inp;
        qw1[i] = (int8_t*)alloc((size_t)p32(pl) * p64(inp));
        aw1[i] = slot();
        packW(w1f[i], pl, inp, p32(pl), p64(inp), qw1[i], aw1[i]);
        qw2[i] = (int8_t*)alloc((size_t)p32(pl) * 9);
        aw2[i] = slot();
        packW(w2f[i], pl, 9, p32(pl), 9, qw2[i], aw2[i]);
        qw3[i] = (int8_t*)alloc((size_t)p32(outp) * p64(pl));
        aw3[i] = slot();
        packW(w3f[i], outp, pl, p32(outp), p64(pl), qw3[i], aw3[i]);
        if (wsf[i]) {
            qws[i] = (int8_t*)alloc((size_t)p32(outp) * p64(inp));
            aws[i] = slot();
            packW(wsf[i], outp, inp, p32(outp), p64(inp), qws[i], aws[i]);
        } else { qws[i] = nullptr; aws[i] = nullptr; }
    }
    int8_t* qc2 = (int8_t*)alloc((size_t)1280 * 320);
    float* ac2 = slot();
    packW(Wc2, 1280, 320, 1280, 320, qc2, ac2);

    h16* F_A = (h16*)alloc(4194304ULL * 2);
    h16* F_B = (h16*)alloc(4194304ULL * 2);
    h16* H4F = (h16*)alloc(4194304ULL * 2);
    h16* H1F = (h16*)alloc(20971520ULL * 2);
    h16* H2F = (h16*)alloc(20971520ULL * 2);
    int8_t* OUTQ = (int8_t*)alloc(8388608ULL);
    int8_t* H1Q = (int8_t*)alloc(25165824ULL);
    int8_t* H2Q = (int8_t*)alloc(25165824ULL);

    auto gemm = [&](const int8_t* A, const int8_t* B, h16* O, const h16* R,
                    long long M, int N32, int K64, int ldO, int ldR,
                    const float* aA, const float* aB, int relu) {
        long long waves = (M / 32) * (long long)(N32 / 32);
        long long thr = waves * 32;
        k_gemm_iu8<<<dim3((unsigned)((thr + 255) / 256)), dim3(256), 0, stream>>>(
            A, B, O, R, (int)M, N32, K64, ldO, ldR, aA, aB, relu);
    };
    auto quantAct = [&](const h16* F, long long M, int C, int ldF, int8_t* Q,
                        int ldQ) -> const float* {
        float* am = slot();
        amax16(F, M * (long long)ldF, am);
        long long tot = M * (long long)ldQ;
        k_quant<<<dim3((unsigned)((tot + 255) / 256)), dim3(256), 0, stream>>>(
            F, M, C, ldF, Q, ldQ, am);
        return am;
    };

    h16* outF = F_A;
    h16* altF = F_B;
    k_conv1<<<dim3(131072 * 32 / 256), dim3(256), 0, stream>>>(X, qc1, ac1, outF);
    int Hc = 32;
    long long Mcur = 128LL * 32 * 32;
    const float* aout = quantAct(outF, Mcur, 32, 32, OUTQ, 64);

    for (int i = 0; i < 17; ++i) {
        int inp = SPEC[i][0], outp = SPEC[i][1], ex = SPEC[i][2], st = SPEC[i][3];
        int pl = ex * inp;
        int Hin = Hc, Hout = Hin / st;
        long long Min = Mcur, Mout = 128LL * Hout * Hout;

        gemm(OUTQ, qw1[i], H1F, nullptr, Min, p32(pl), p64(inp), p32(pl), 0,
             aout, aw1[i], 1);
        const float* ah1 = quantAct(H1F, Min, pl, p32(pl), H1Q, p64(pl));

        long long tdw = Mout * (long long)p32(pl);
        k_dwconv<<<dim3((unsigned)((tdw + 255) / 256)), dim3(256), 0, stream>>>(
            H1Q, qw2[i], H2F, Hin, pl, p32(pl), p64(pl), st, ah1, aw2[i]);
        const float* ah2 = quantAct(H2F, Mout, pl, p32(pl), H2Q, p64(pl));

        const h16* res = nullptr;
        int ldR = 0;
        if (st == 1) {
            if (inp != outp) {
                gemm(OUTQ, qws[i], H4F, nullptr, Min, p32(outp), p64(inp),
                     p32(outp), 0, aout, aws[i], 0);
                res = H4F; ldR = p32(outp);
            } else { res = outF; ldR = p32(inp); }
        }

        gemm(H2Q, qw3[i], altF, res, Mout, p32(outp), p64(pl), p32(outp), ldR,
             ah2, aw3[i], 0);

        h16* t = outF; outF = altF; altF = t;
        Mcur = Mout; Hc = Hout;
        aout = quantAct(outF, Mcur, outp, p32(outp), OUTQ, p64(outp));
    }

    gemm(OUTQ, qc2, altF, nullptr, Mcur, 1280, 320, 1280, 0, aout, ac2, 1);

    k_poolfc<<<dim3(128), dim3(256), 0, stream>>>(altF, FCW, FCB, (float*)d_out);
}
